// KTBCrossAttention_6571299963395
// MI455X (gfx1250) — matrix operand order repaired and checked by reading, not compiled, not run on hardware
//
#include <hip/hip_runtime.h>

// ---------------------------------------------------------------------------
// KTB cross attention for MI455X (gfx1250), wave32 + WMMA bf16.
//   DIM=768, HEADS=12, HEAD_DIM=64, B=16, N=4096, SCALE=0.125
// Pipeline: cvt_wt -> kv_gemm(rgb) -> ctx(rgb) -> kv_gemm(depth) -> ctx(depth)
//           -> out(rgb uses ctx_depth) -> out(depth uses ctx_rgb)
// KV is produced TRANSPOSED (KVt[b][c][n], bf16) so ctx_kernel loads WMMA
// fragments directly from global with 16B accesses and no LDS transposes.
// kv_gemm is double-buffered in LDS with register-prefetch of the next tile.
// ---------------------------------------------------------------------------

#define DIMC   768
#define NCOL   1536
#define HEADS  12
#define NSEQ   4096
#define BATCH  16
#define HD     64
#define SCALEF 0.125f
#define KSTEPS (DIMC / 32)  // 24

typedef __bf16 bf16_t;
typedef __attribute__((ext_vector_type(16))) __bf16 v16bf;
typedef __attribute__((ext_vector_type(8)))  __bf16 v8bf;
typedef __attribute__((ext_vector_type(8)))  float  v8f;

__device__ __forceinline__ v8f wmma_bf16(v16bf a, v16bf b, v8f c) {
  return __builtin_amdgcn_wmma_f32_16x16x32_bf16(
      false, a, false, b, (short)0, c, false, false);
}

__device__ __forceinline__ v16bf pack16(v8bf lo, v8bf hi) {
  v16bf r;
#pragma unroll
  for (int i = 0; i < 8; ++i) { r[i] = lo[i]; r[i + 8] = hi[i]; }
  return r;
}

// ---------------------------------------------------------------------------
// Kernel 0: W[k][n] fp32  ->  Wt[n][k] bf16   (transpose via 32x32 LDS tile)
// ---------------------------------------------------------------------------
__global__ __launch_bounds__(256) void cvt_wt_kernel(
    const float* __restrict__ W, bf16_t* __restrict__ Wt) {
  __shared__ float tile[32][33];
  const int kt = blockIdx.x * 32;  // k tile (24 tiles)
  const int nt = blockIdx.y * 32;  // n tile (48 tiles)
  const int tx = threadIdx.x & 31;
  const int ty = threadIdx.x >> 5;  // 0..7
#pragma unroll
  for (int r = 0; r < 4; ++r) {  // coalesced read along n
    int k = ty + r * 8;
    tile[k][tx] = W[(size_t)(kt + k) * NCOL + nt + tx];
  }
  __syncthreads();
#pragma unroll
  for (int r = 0; r < 4; ++r) {  // coalesced write along k
    int n = ty + r * 8;
    Wt[(size_t)(nt + n) * DIMC + kt + tx] = (bf16_t)tile[tx][n];
  }
}

// ---------------------------------------------------------------------------
// Kernel 1: KVt[b][c][n] = (X @ W)^T per batch  (M=65536, K=768, N=1536)
// WG 256 (8 waves), tile 128M x 128N, k-step 32, LDS double-buffered with
// register prefetch of the next tile. Wave tile 32M x 64N = 2x4 WMMA accs.
// ---------------------------------------------------------------------------
__global__ __launch_bounds__(256) void kv_gemm_kernel(
    const float* __restrict__ X, const bf16_t* __restrict__ Wtg,
    bf16_t* __restrict__ KVt) {
  __shared__ __align__(16) bf16_t Xs[2][128][40];  // [m][k], stride 80B
  __shared__ __align__(16) bf16_t Ws[2][128][40];  // [ncol][k], stride 80B

  const int tid   = threadIdx.x;
  const int lane  = tid & 31;
  const int wid   = tid >> 5;
  const int l16   = lane & 15;
  const int khalf = lane >> 4;
  const int wm    = (wid & 3) * 32;
  const int wn    = (wid >> 2) * 64;
  const size_t mbase = (size_t)blockIdx.y * 128;
  const int    nbase = blockIdx.x * 128;

  float4 xr[4];  // staged X chunk (fp32)
  v8bf   wr[2];  // staged W chunk (bf16, already transposed in global)

  auto load_tile = [&](int k0) {
#pragma unroll
    for (int r = 0; r < 4; ++r) {
      int idx = tid + r * 256;          // 0..1023 -> 128 rows x 8 float4
      int row = idx >> 3;
      int kq  = (idx & 7) << 2;
      xr[r] = *(const float4*)(X + (mbase + row) * DIMC + k0 + kq);
    }
#pragma unroll
    for (int r = 0; r < 2; ++r) {
      int idx = tid + r * 256;          // 0..511 -> 128 rows x 4 v8bf
      int row = idx >> 2;
      int kc  = (idx & 3) << 3;
      wr[r] = *(const v8bf*)(Wtg + (size_t)(nbase + row) * DIMC + k0 + kc);
    }
  };
  auto store_tile = [&](int buf) {
#pragma unroll
    for (int r = 0; r < 4; ++r) {
      int idx = tid + r * 256;
      int row = idx >> 3;
      int kq  = (idx & 7) << 2;
      Xs[buf][row][kq + 0] = (bf16_t)xr[r].x;  // -> v_cvt_pk_bf16 + ds_store_b64
      Xs[buf][row][kq + 1] = (bf16_t)xr[r].y;
      Xs[buf][row][kq + 2] = (bf16_t)xr[r].z;
      Xs[buf][row][kq + 3] = (bf16_t)xr[r].w;
    }
#pragma unroll
    for (int r = 0; r < 2; ++r) {
      int idx = tid + r * 256;
      int row = idx >> 2;
      int kc  = (idx & 3) << 3;
      *(v8bf*)&Ws[buf][row][kc] = wr[r];       // ds_store_b128
    }
  };

  v8f acc[2][4] = {};

  load_tile(0);
  store_tile(0);
  __syncthreads();

  for (int i = 0; i < KSTEPS; ++i) {
    const int cur = i & 1;
    if (i + 1 < KSTEPS) load_tile((i + 1) * 32);  // global loads in flight

    v16bf af[2], bfr[4];
#pragma unroll
    for (int mt = 0; mt < 2; ++mt) {
      int row = wm + mt * 16 + l16;
      af[mt] = pack16(*(const v8bf*)&Xs[cur][row][khalf * 8],
                      *(const v8bf*)&Xs[cur][row][16 + khalf * 8]);
    }
#pragma unroll
    for (int nt = 0; nt < 4; ++nt) {
      int col = wn + nt * 16 + l16;
      bfr[nt] = pack16(*(const v8bf*)&Ws[cur][col][khalf * 8],
                       *(const v8bf*)&Ws[cur][col][16 + khalf * 8]);
    }
#pragma unroll
    for (int mt = 0; mt < 2; ++mt)
#pragma unroll
      for (int nt = 0; nt < 4; ++nt)
        acc[mt][nt] = wmma_bf16(af[mt], bfr[nt], acc[mt][nt]);

    __syncthreads();                     // readers of buf 'cur^1' are done
    if (i + 1 < KSTEPS) {
      store_tile(cur ^ 1);               // loadcnt wait lands after the WMMAs
      __syncthreads();
    }
  }

  // ---- store transposed: lane holds col c, rows khalf*8..+7 = consecutive n
  const size_t bo   = (mbase / NSEQ) * (size_t)NCOL * NSEQ;
  const int    nloc = (int)(mbase % NSEQ);
#pragma unroll
  for (int mt = 0; mt < 2; ++mt)
#pragma unroll
    for (int nt = 0; nt < 4; ++nt) {
      int c = nbase + wn + nt * 16 + l16;
      int n = nloc + wm + mt * 16 + khalf * 8;
      v8bf pk;
#pragma unroll
      for (int v = 0; v < 8; ++v) pk[v] = (bf16_t)acc[mt][nt][v];
      *(v8bf*)(KVt + bo + (size_t)c * NSEQ + n) = pk;  // 16B packed store
    }
}

// ---------------------------------------------------------------------------
// Kernel 2: ctx = softmax_d( SCALE * K^T V ) per (b,h); writes ctx^T bf16.
// WG 256 (8 waves); fragments loaded straight from transposed KVt (16B).
// No LDS in the main loop; deterministic fixed-order wave reduction at end.
// ---------------------------------------------------------------------------
__global__ __launch_bounds__(256) void ctx_kernel(
    const bf16_t* __restrict__ KVt, bf16_t* __restrict__ ctxT) {
  __shared__ __align__(16) float p0[64][65];
  __shared__ __align__(16) float p1[64][65];

  const int tid   = threadIdx.x;
  const int lane  = tid & 31;
  const int wid   = tid >> 5;
  const int l16   = lane & 15;
  const int khalf = lane >> 4;
  const int b     = blockIdx.x / HEADS;
  const int h     = blockIdx.x % HEADS;
  const bf16_t* Kp = KVt + ((size_t)b * NCOL + h * HD) * NSEQ;  // Kt[d][n]
  const bf16_t* Vp = Kp + (size_t)768 * NSEQ;                   // Vt[e][n]

  v8f acc[4][4] = {};

  for (int n0 = wid * 32; n0 < NSEQ; n0 += 256) {
    v16bf af[4], bfr[4];
#pragma unroll
    for (int dt = 0; dt < 4; ++dt) {  // A = K^T : row d, K-dim = n
      const bf16_t* kp = Kp + (size_t)(dt * 16 + l16) * NSEQ + n0;
      af[dt] = pack16(*(const v8bf*)(kp + khalf * 8),
                      *(const v8bf*)(kp + 16 + khalf * 8));
    }
#pragma unroll
    for (int et = 0; et < 4; ++et) {  // B = V : b[k] = V[n][e] = Vt[e][n]
      const bf16_t* vp = Vp + (size_t)(et * 16 + l16) * NSEQ + n0;
      bfr[et] = pack16(*(const v8bf*)(vp + khalf * 8),
                       *(const v8bf*)(vp + 16 + khalf * 8));
    }
#pragma unroll
    for (int dt = 0; dt < 4; ++dt)
#pragma unroll
      for (int et = 0; et < 4; ++et)
        acc[dt][et] = wmma_bf16(af[dt], bfr[et], acc[dt][et]);
  }

  // ---- deterministic reduction: (w0,w1) store; (w2,w3),(w4,w5),(w6,w7) add
  float(*pb)[65] = (wid & 1) ? p1 : p0;
#define CTX_RED(OPEQ)                                                     \
  {                                                                       \
    _Pragma("unroll") for (int dt = 0; dt < 4; ++dt)                      \
    _Pragma("unroll") for (int et = 0; et < 4; ++et)                      \
    _Pragma("unroll") for (int v = 0; v < 8; ++v)                         \
        pb[dt * 16 + khalf * 8 + v][et * 16 + l16] OPEQ acc[dt][et][v];   \
  }
  if (wid < 2) CTX_RED(=)
  __syncthreads();
  if (wid == 2 || wid == 3) CTX_RED(+=)
  __syncthreads();
  if (wid == 4 || wid == 5) CTX_RED(+=)
  __syncthreads();
  if (wid >= 6) CTX_RED(+=)
  __syncthreads();
#undef CTX_RED

  // ---- column-wise softmax over d (axis=-2), write ctx^T[e][d] bf16
  if (tid < 64) {
    const int e = tid;
    float mx = -3.4e38f;
    for (int d = 0; d < 64; ++d) {
      float s = (p0[d][e] + p1[d][e]) * SCALEF;
      mx = fmaxf(mx, s);
    }
    float sum = 0.f;
    for (int d = 0; d < 64; ++d)
      sum += __expf((p0[d][e] + p1[d][e]) * SCALEF - mx);
    float inv = 1.f / sum;
    bf16_t* dst = ctxT + ((size_t)blockIdx.x * 64 + e) * 64;
    for (int d = 0; d < 64; ++d)
      dst[d] = (bf16_t)(__expf((p0[d][e] + p1[d][e]) * SCALEF - mx) * inv);
  }
}

// ---------------------------------------------------------------------------
// Kernel 3: out[b,n,h*64+e] = sum_d q[b,n,h*64+d] * ctx_cross[d][e], fp32 out.
// grid = (N/128, B*HEADS); WG 256 (8 waves); wave = 16 n-rows x 64 cols.
// ---------------------------------------------------------------------------
__global__ __launch_bounds__(256) void out_kernel(
    const float* __restrict__ X, const bf16_t* __restrict__ ctxTc,
    float* __restrict__ OUT) {
  const int tid   = threadIdx.x;
  const int lane  = tid & 31;
  const int wid   = tid >> 5;
  const int l16   = lane & 15;
  const int khalf = lane >> 4;
  const int bh = blockIdx.y;
  const int b  = bh / HEADS;
  const int h  = bh % HEADS;
  const int n0 = blockIdx.x * 128 + wid * 16;

  v8f acc[4] = {};
#pragma unroll
  for (int k0 = 0; k0 < 64; k0 += 32) {
    const float* xp = X + ((size_t)b * NSEQ + n0 + l16) * DIMC + h * HD + k0;
    float4 f0 = *(const float4*)(xp + khalf * 8);
    float4 f1 = *(const float4*)(xp + khalf * 8 + 4);
    float4 f2 = *(const float4*)(xp + 16 + khalf * 8);
    float4 f3 = *(const float4*)(xp + 16 + khalf * 8 + 4);
    float fx[16] = {f0.x, f0.y, f0.z, f0.w, f1.x, f1.y, f1.z, f1.w,
                    f2.x, f2.y, f2.z, f2.w, f3.x, f3.y, f3.z, f3.w};
    v16bf af;
#pragma unroll
    for (int i = 0; i < 16; ++i) af[i] = (bf16_t)fx[i];

#pragma unroll
    for (int et = 0; et < 4; ++et) {  // B = ctx[d][e] from ctx^T[e][d]
      int e = et * 16 + l16;
      const bf16_t* cp = ctxTc + ((size_t)bh * 64 + e) * 64 + k0 + khalf * 8;
      v16bf bfr = pack16(*(const v8bf*)cp, *(const v8bf*)(cp + 16));
      acc[et] = wmma_bf16(af, bfr, acc[et]);
    }
  }
#pragma unroll
  for (int et = 0; et < 4; ++et) {
    float* op = OUT + ((size_t)b * NSEQ + n0 + khalf * 8) * DIMC + h * HD +
                et * 16 + l16;
#pragma unroll
    for (int v = 0; v < 8; ++v) op[(size_t)v * DIMC] = acc[et][v];
  }
}

// ---------------------------------------------------------------------------
// Launch. Workspace: Wt bf16 (2 x 1536 x 768) | KVt bf16 (16 x 1536 x 4096,
// reused rgb->depth) | ctxT bf16 (2 x 192 x 64 x 64). ~208 MB total.
// ---------------------------------------------------------------------------
extern "C" void kernel_launch(void* const* d_in, const int* in_sizes, int n_in,
                              void* d_out, int out_size, void* d_ws,
                              size_t ws_size, hipStream_t stream) {
  const float* rgb = (const float*)d_in[0];
  const float* dep = (const float*)d_in[1];
  const float* Wr  = (const float*)d_in[2];
  const float* Wd  = (const float*)d_in[3];

  float* out_rgb = (float*)d_out;
  float* out_dep = out_rgb + (size_t)BATCH * NSEQ * DIMC;

  const size_t wElems = (size_t)DIMC * NCOL;  // 1,179,648
  bf16_t* Wt   = (bf16_t*)d_ws;               // 2 * wElems (transposed bf16)
  bf16_t* kvt  = Wt + 2 * wElems;             // 16 * 1536 * 4096
  bf16_t* ctxT = kvt + (size_t)BATCH * NCOL * NSEQ;
  const size_t ctxStride = (size_t)BATCH * HEADS * 64 * 64;

  dim3 gCvt(DIMC / 32, NCOL / 32);               // (24, 48)
  cvt_wt_kernel<<<gCvt, 256, 0, stream>>>(Wr, Wt);
  cvt_wt_kernel<<<gCvt, 256, 0, stream>>>(Wd, Wt + wElems);

  dim3 gGemm(NCOL / 128, (BATCH * NSEQ) / 128);  // (12, 512)
  dim3 gCtx(BATCH * HEADS);                      // 192
  dim3 gOut(NSEQ / 128, BATCH * HEADS);          // (32, 192)

  // rgb: KV^T then ctx_rgb (KVt buffer reused afterwards)
  kv_gemm_kernel<<<gGemm, 256, 0, stream>>>(rgb, Wt, kvt);
  ctx_kernel<<<gCtx, 256, 0, stream>>>(kvt, ctxT);
  // depth: KV^T then ctx_depth
  kv_gemm_kernel<<<gGemm, 256, 0, stream>>>(dep, Wt + wElems, kvt);
  ctx_kernel<<<gCtx, 256, 0, stream>>>(kvt, ctxT + ctxStride);
  // cross outputs
  out_kernel<<<gOut, 256, 0, stream>>>(rgb, ctxT + ctxStride, out_rgb);
  out_kernel<<<gOut, 256, 0, stream>>>(dep, ctxT, out_dep);
}
